// DynamicFFTConv1d_90821378441578
// MI455X (gfx1250) — matrix operand order repaired and checked by reading, not compiled, not run on hardware
//
#include <hip/hip_runtime.h>
#include <stdint.h>
#include <stddef.h>

typedef __attribute__((ext_vector_type(16))) _Float16 v16h;
typedef __attribute__((ext_vector_type(8)))  _Float16 v8h;
typedef __attribute__((ext_vector_type(8)))  float    v8f;
typedef __attribute__((ext_vector_type(4)))  unsigned int u32x4;
typedef __attribute__((ext_vector_type(8)))  int      i32x8;
typedef __attribute__((ext_vector_type(4)))  int      i32x4;

namespace {
constexpr int   kT    = 32768;
constexpr int   kB    = 8;
constexpr int   kIn   = 16;
constexpr int   kOut  = 16;
constexpr int   kCond = 128;
constexpr int   kNF   = 513;              // STFT frames
constexpr int   kNSig = kB * kIn;         // 128
constexpr int   kRows = kNSig * kNF;      // 65664 (= 4104 * 16 exactly)
constexpr int   kM2   = 2 * kOut * kIn * 65;   // 33280
constexpr int   kTP   = kT + 128;         // zero-padded conditioning row (32896)
constexpr float kPi   = 3.14159265358979323846f;

// ---- workspace layout (all sizes 256B-aligned) ----
constexpr size_t OFF_CONDP  = 0;                                   // f16 (B,128,T+128) zero-padded
constexpr size_t SZ_CONDP   = (size_t)kB * kCond * kTP * 2;
constexpr size_t OFF_W1     = OFF_CONDP + SZ_CONDP;                // f16 (256,16384)
constexpr size_t SZ_W1      = (size_t)256 * 16384 * 2;
constexpr size_t OFF_W2     = OFF_W1 + SZ_W1;                      // f16 (33280,256)
constexpr size_t SZ_W2      = (size_t)kM2 * 256 * 2;
constexpr size_t OFF_BAS1   = OFF_W2 + SZ_W2;                      // f16 transposed (144,128) fwd DFT
constexpr size_t SZ_BAS1    = (size_t)144 * 128 * 2;
constexpr size_t OFF_BAS2   = OFF_BAS1 + SZ_BAS1;                  // f16 transposed (128,160) inv DFT
constexpr size_t SZ_BAS2    = (size_t)128 * 160 * 2;
constexpr size_t OFF_FRIN   = OFF_BAS2 + SZ_BAS2;                  // f16 (rows,128) windowed frames
constexpr size_t SZ_FRIN    = (size_t)kRows * 128 * 2;
constexpr size_t OFF_X      = OFF_FRIN + SZ_FRIN;                  // f16 [b][t][i][130]
constexpr size_t SZ_X       = (size_t)kRows * 130 * 2;
constexpr size_t OFF_H      = OFF_X + SZ_X;                        // f32 [b*513+t][256]
constexpr size_t SZ_H       = (size_t)kB * kNF * 256 * 4;
constexpr size_t OFF_Y16    = OFF_H + SZ_H;                        // f16 (rows,160) spectrum for iSTFT
constexpr size_t SZ_Y16     = (size_t)kRows * 160 * 2;
constexpr size_t OFF_FROUT  = OFF_Y16 + SZ_Y16;                    // f32 (rows,128) time frames
}

// ---------------- WMMA fragment helpers (contiguous k = 16*(lane/16) + s in A and B, wave32) -------------
__device__ __forceinline__ v16h load_a_rowmajor(const _Float16* p, int ld) {
  int lane = threadIdx.x & 31;
  const _Float16* rp = p + (size_t)(lane & 15) * ld;
  int kbase = (lane < 16) ? 0 : 16;
  v16h a;
#pragma unroll
  for (int v = 0; v < 8; ++v) {
    int k0 = kbase + (v << 1);
    a[2 * v]     = rp[k0];
    a[2 * v + 1] = rp[k0 + 1];
  }
  return a;
}

// B fragment from [n][k]-transposed storage: each lane reads 16 contiguous f16.
__device__ __forceinline__ v16h load_b_contig(const _Float16* p) {
  union { v16h v; v8h h[2]; } u;
  u.h[0] = *(const v8h*)p;
  u.h[1] = *(const v8h*)(p + 8);
  return u.v;
}
__device__ __forceinline__ v16h load_b_trans(const _Float16* base, int ld) {
  int lane = threadIdx.x & 31;
  return load_b_contig(base + (size_t)(lane & 15) * ld + ((lane < 16) ? 0 : 16));
}

__device__ __forceinline__ float hannw(int n) {
  return 0.5f * (1.0f - cosf((float)n * (kPi / 64.0f)));
}

// ---------------- setup kernels ----------------
__global__ void cvt_f16_kernel(const float* __restrict__ src, _Float16* __restrict__ dst, int n) {
  int i = blockIdx.x * blockDim.x + threadIdx.x;
  if (i < n) dst[i] = (_Float16)src[i];
}

// conditioning -> f16 with 64-sample zero pad on both sides of each channel row
__global__ void cvt_cond_kernel(const float* __restrict__ src, _Float16* __restrict__ dst) {
  int i = blockIdx.x * blockDim.x + threadIdx.x;
  if (i >= kB * kCond * kTP) return;
  int row = i / kTP, sp = i - row * kTP;
  int s = sp - 64;
  float v = (s >= 0 && s < kT) ? src[(size_t)row * kT + s] : 0.f;
  dst[i] = (_Float16)v;
}

__global__ void zero32_kernel(uint32_t* __restrict__ dst, int n) {
  int i = blockIdx.x * blockDim.x + threadIdx.x;
  if (i < n) dst[i] = 0u;
}

// basis tables generated directly in [n][k]-transposed order
__global__ void make_basis_kernel(_Float16* __restrict__ b1t, _Float16* __restrict__ b2t) {
  int idx = blockIdx.x * blockDim.x + threadIdx.x;
  if (idx < 144 * 128) {                 // fwd: b1t[col][n], col 0..64 Re, 65..129 Im, pad
    int col = idx / 128, n = idx % 128;
    float v = 0.f;
    if (col < 65)       { int ph = (n * col) & 127;        v =  cosf(ph * (kPi / 64.f)); }
    else if (col < 130) { int ph = (n * (col - 65)) & 127; v = -sinf(ph * (kPi / 64.f)); }
    b1t[idx] = (_Float16)v;
  }
  if (idx < 128 * 160) {                 // inv: b2t[n][r], r 0..64 Re, 65..129 Im, pad
    int n = idx / 160, r = idx % 160;
    float v = 0.f;
    if (r < 65) {
      int fb = r; float c = (fb == 0 || fb == 64) ? 1.f : 2.f;
      int ph = (n * fb) & 127; v = c * cosf(ph * (kPi / 64.f)) * (1.f / 128.f);
    } else if (r < 130) {
      int fb = r - 65; float c = (fb == 0 || fb == 64) ? 1.f : 2.f;
      int ph = (n * fb) & 127; v = -c * sinf(ph * (kPi / 64.f)) * (1.f / 128.f);
    }
    b2t[idx] = (_Float16)v;
  }
}

// reflect-pad + Hann window -> f16 frame matrix (rows, 128)
__global__ void stft_gather_kernel(const float* __restrict__ x, _Float16* __restrict__ frin) {
  int idx = blockIdx.x * blockDim.x + threadIdx.x;
  if (idx >= kRows * 128) return;
  int n   = idx & 127;
  int rf  = idx >> 7;
  int fr  = rf % kNF;
  int sig = rf / kNF;
  int s = fr * 64 + n - 64;
  if (s < 0) s = -s;
  if (s >= kT) s = 2 * kT - 2 - s;
  frin[idx] = (_Float16)(x[(size_t)sig * kT + s] * hannw(n));
}

// ---------------- STFT GEMM: (65664 x 128) * (128 x 130) -> X f16 --------------
__global__ void __launch_bounds__(256)
gemm_stft_kernel(const _Float16* __restrict__ frin, const _Float16* __restrict__ bas1t,
                 _Float16* __restrict__ Xbuf) {
  int lane = threadIdx.x & 31;
  int wt = blockIdx.x * 8 + (threadIdx.x >> 5);   // 4104*9 tiles exactly
  int mt = wt / 9, nt = wt % 9;
  v8f acc = {};
#pragma unroll
  for (int kk = 0; kk < 4; ++kk) {
    v16h a = load_a_rowmajor(frin + (size_t)(mt * 16) * 128 + kk * 32, 128);
    v16h b = load_b_trans(bas1t + (size_t)(nt * 16) * 128 + kk * 32, 128);
    acc = __builtin_amdgcn_wmma_f32_16x16x32_f16(false, a, false, b, (short)0, acc, false, false);
  }
  int mo = (lane >> 4) << 3, nn = lane & 15;
  int col = nt * 16 + nn;
  if (col < 130) {
#pragma unroll
    for (int r = 0; r < 8; ++r) {
      int row = mt * 16 + mo + r;
      int sig = row / kNF, fr = row - sig * kNF;
      int bb = sig >> 4, ii = sig & 15;
      Xbuf[(((size_t)bb * kNF + fr) * 16 + ii) * 130 + col] = (_Float16)acc[r];
    }
  }
}

// ------------- conv1 GEMM, on-the-fly im2col from padded f16: M=256 N=4104 K=16384
__global__ void __launch_bounds__(256)
gemm_conv1_kernel(const _Float16* __restrict__ w1h, const _Float16* __restrict__ condp,
                  const float* __restrict__ b1, float* __restrict__ hbuf) {
  int lane = threadIdx.x & 31;
  int wt = blockIdx.x * 8 + (threadIdx.x >> 5);   // 16*257 tiles exactly
  int mt = wt & 15, nt = wt >> 4;
  int col = nt * 16 + (lane & 15);
  bool cvalid = col < kB * kNF;
  int ccol = cvalid ? col : (kB * kNF - 1);       // clamp: safe reads, stores suppressed
  int bb = ccol / kNF;
  int t  = ccol - bb * kNF;
  int kb = (lane < 16) ? 0 : 16;
  const _Float16* condb = condp + (size_t)bb * kCond * kTP + (size_t)t * 64;
  v8f acc = {};
  for (int kk = 0; kk < 512; ++kk) {
    v16h a = load_a_rowmajor(w1h + (size_t)(mt * 16) * 16384 + kk * 32, 16384);
    __builtin_prefetch(w1h + (size_t)(mt * 16) * 16384 + (kk + 4) * 32, 0, 1);
    int k0 = kk * 32 + kb;                        // 16-run never crosses a channel
    int c = k0 >> 7, kw = k0 & 127;
    v16h bf = load_b_contig(condb + (size_t)c * kTP + kw);
    acc = __builtin_amdgcn_wmma_f32_16x16x32_f16(false, a, false, bf, (short)0, acc, false, false);
  }
  int mo = (lane >> 4) << 3;
  if (cvalid) {
#pragma unroll
    for (int r = 0; r < 8; ++r) {
      int row = mt * 16 + mo + r;
      hbuf[(size_t)col * 256 + row] = acc[r] + b1[row];
    }
  }
}

// ------- fused conv2 GEMM + spectral filter apply (filters never hit HBM) -------
__global__ void __launch_bounds__(256)
fused_conv2_einsum_kernel(const _Float16* __restrict__ w2h, const float* __restrict__ b2,
                          const float* __restrict__ hbuf, const _Float16* __restrict__ Xbuf,
                          _Float16* __restrict__ y16) {
  __shared__ _Float16 hT[16 * 264];   // [t'][c], 16B row pad for bank rotation
  __shared__ _Float16 Xs[16 * 2080];  // [t'][i*130 + p*65 + fb]
  __shared__ float    Yo[1040];       // [fb][t'] accumulator (per p,o)
  int tid = threadIdx.x, lane = tid & 31, wid = tid >> 5;
  int bidx = blockIdx.x / 33;
  int t0   = (blockIdx.x % 33) * 16;

  // ---- X tile: contiguous 66,560B global -> LDS via Tensor Data Mover ----
  const _Float16* xsrc = Xbuf + ((size_t)bidx * kNF + t0) * 2080;
#if __has_builtin(__builtin_amdgcn_tensor_load_to_lds) && __has_builtin(__builtin_amdgcn_s_wait_tensorcnt)
  if (tid == 0) {
    uint64_t ga = (uint64_t)(uintptr_t)xsrc;
    uint32_t la = (uint32_t)(uintptr_t)(void*)Xs;       // LDS aperture: low 32 bits = LDS offset
    u32x4 g0 = { 1u,                                    // count=1 (valid user descriptor)
                 la,                                     // lds_addr
                 (uint32_t)ga,                           // global_addr[31:0]
                 (uint32_t)((ga >> 32) & 0x1FFFFFFu) | (2u << 30) };  // addr[56:32] | type=2
    const uint32_t nUnits = 16u * 2080u * 2u / 8u;       // 8320 x 8B units (1-D tile)
    i32x8 g1 = { (int)(3u << 16),                        // data_size=8B
                 (int)((nUnits & 0xFFFFu) << 16),        // tensor_dim0[15:0]
                 (int)(nUnits >> 16),                    // tensor_dim0[31:16]
                 (int)((nUnits & 0xFFFFu) << 16),        // tile_dim0
                 0, 0, 0, 0 };
    i32x4 gz4 = { 0, 0, 0, 0 };
    i32x8 gz8 = { 0, 0, 0, 0, 0, 0, 0, 0 };
    // 6-arg form (clang-23 / therock-10.0 lane): groups 2/3 disabled, aux group zero
    __builtin_amdgcn_tensor_load_to_lds(g0, g1, gz4, gz4, gz8, 0);
  }
#else
  {
    const uint32_t* s32 = (const uint32_t*)xsrc;
    uint32_t* d32 = (uint32_t*)Xs;
    for (int idx = tid; idx < 16 * 1040; idx += 256) d32[idx] = s32[idx];
  }
#endif

  // ---- h tile (f32 -> f16), transposed [t'][c], zero-padded past frame 512 ----
  for (int idx = tid; idx < 16 * 256; idx += 256) {
    int tt = idx >> 8, c = idx & 255, t = t0 + tt;
    float v = (t < kNF) ? hbuf[((size_t)bidx * kNF + t) * 256 + c] : 0.f;
    hT[tt * 264 + c] = (_Float16)v;
  }
#if __has_builtin(__builtin_amdgcn_tensor_load_to_lds) && __has_builtin(__builtin_amdgcn_s_wait_tensorcnt)
  if (tid == 0) __builtin_amdgcn_s_wait_tensorcnt((short)0);
#endif
  __syncthreads();

  int mo = (lane >> 4) << 3, nn = lane & 15;
  for (int po = 0; po < 32; ++po) {
    int p = po & 1, o = po >> 1;
    int base = (p * 16 + o) * 1040;           // contiguous W2 row chunk
    for (int idx = tid; idx < 1040; idx += 256) Yo[idx] = 0.f;
    __syncthreads();
    for (int mt = wid; mt < 65; mt += 8) {    // wave-uniform tiling, EXEC stays full
      v8f acc = {};
#pragma unroll
      for (int kk = 0; kk < 8; ++kk) {
        v16h a = load_a_rowmajor(w2h + (size_t)(base + mt * 16) * 256 + kk * 32, 256);
        v16h b = load_b_trans(hT + kk * 32, 264);
        acc = __builtin_amdgcn_wmma_f32_16x16x32_f16(false, a, false, b, (short)0, acc, false, false);
      }
#pragma unroll
      for (int r = 0; r < 8; ++r) {
        int ml = mt * 16 + mo + r;            // = i*65 + fb
        int i = ml / 65, fb = ml - i * 65;
        float f  = acc[r] + b2[base + ml];
        float xv = (float)Xs[nn * 2080 + i * 130 + p * 65 + fb];
        atomicAdd(&Yo[fb * 16 + nn], f * xv); // ds_add_f32, reduces over i
      }
    }
    __syncthreads();
    for (int idx = tid; idx < 1040; idx += 256) {
      int fb = idx >> 4, tt = idx & 15, t = t0 + tt;
      if (t < kNF)
        y16[((size_t)(bidx * 16 + o) * kNF + t) * 160 + p * 65 + fb] = (_Float16)Yo[idx];
    }
    __syncthreads();
  }
}

// -------------- iSTFT GEMM: (65664 x 160) * (160 x 128) -> time frames ----------
__global__ void __launch_bounds__(256)
gemm_istft_kernel(const _Float16* __restrict__ y16, const _Float16* __restrict__ bas2t,
                  float* __restrict__ frout) {
  int lane = threadIdx.x & 31;
  int wt = blockIdx.x * 8 + (threadIdx.x >> 5);  // 4104*8 tiles exactly
  int mt = wt >> 3, nt = wt & 7;
  v8f acc = {};
#pragma unroll
  for (int kk = 0; kk < 5; ++kk) {
    v16h a = load_a_rowmajor(y16 + (size_t)(mt * 16) * 160 + kk * 32, 160);
    v16h b = load_b_trans(bas2t + (size_t)(nt * 16) * 160 + kk * 32, 160);
    acc = __builtin_amdgcn_wmma_f32_16x16x32_f16(false, a, false, b, (short)0, acc, false, false);
  }
  int mo = (lane >> 4) << 3, nn = lane & 15;
#pragma unroll
  for (int r = 0; r < 8; ++r)
    frout[(size_t)(mt * 16 + mo + r) * 128 + nt * 16 + nn] = acc[r];
}

// ------- overlap-add (exactly 2 contributions/sample), Hann^2 envelope, bias ----
__global__ void finalize_kernel(const float* __restrict__ frout, const float* __restrict__ bias,
                                float* __restrict__ out) {
  int idx = blockIdx.x * blockDim.x + threadIdx.x;
  if (idx >= kB * kOut * kT) return;
  int t = idx & (kT - 1);
  int sig = idx >> 15;                  // = b*16 + o
  int s = t + 64;                       // position in padded OLA buffer
  int f1 = s >> 6, n1 = s & 63;
  int f0 = f1 - 1, n0 = n1 + 64;
  float w0 = hannw(n0), w1 = hannw(n1);
  float v = frout[((size_t)sig * kNF + f0) * 128 + n0] * w0 +
            frout[((size_t)sig * kNF + f1) * 128 + n1] * w1;
  float wsq = w0 * w0 + w1 * w1;
  out[idx] = v / fmaxf(wsq, 1e-11f) + bias[sig & 15];
}

// -------------------------------- host side -------------------------------------
extern "C" void kernel_launch(void* const* d_in, const int* in_sizes, int n_in,
                              void* d_out, int out_size, void* d_ws, size_t ws_size,
                              hipStream_t stream) {
  (void)in_sizes; (void)n_in; (void)out_size; (void)ws_size;
  const float* x    = (const float*)d_in[0];
  const float* cond = (const float*)d_in[1];
  const float* w1   = (const float*)d_in[2];
  const float* b1   = (const float*)d_in[3];
  const float* w2   = (const float*)d_in[4];
  const float* b2   = (const float*)d_in[5];
  const float* bias = (const float*)d_in[6];
  float* out = (float*)d_out;
  char* ws = (char*)d_ws;

  _Float16* condp = (_Float16*)(ws + OFF_CONDP);
  _Float16* w1h   = (_Float16*)(ws + OFF_W1);
  _Float16* w2h   = (_Float16*)(ws + OFF_W2);
  _Float16* bas1t = (_Float16*)(ws + OFF_BAS1);
  _Float16* bas2t = (_Float16*)(ws + OFF_BAS2);
  _Float16* frin  = (_Float16*)(ws + OFF_FRIN);
  _Float16* Xbuf  = (_Float16*)(ws + OFF_X);
  float*    hbuf  = (float*)   (ws + OFF_H);
  _Float16* y16   = (_Float16*)(ws + OFF_Y16);
  float*    frout = (float*)   (ws + OFF_FROUT);

  const int THR = 256;
  int nCondP = kB * kCond * kTP;
  cvt_cond_kernel<<<(nCondP + THR - 1) / THR, THR, 0, stream>>>(cond, condp);
  int nW1 = 256 * 16384;
  cvt_f16_kernel<<<(nW1 + THR - 1) / THR, THR, 0, stream>>>(w1, w1h, nW1);
  int nW2 = kM2 * 256;
  cvt_f16_kernel<<<(nW2 + THR - 1) / THR, THR, 0, stream>>>(w2, w2h, nW2);
  int nZ = (int)(SZ_Y16 / 4);
  zero32_kernel<<<(nZ + THR - 1) / THR, THR, 0, stream>>>((uint32_t*)y16, nZ);
  make_basis_kernel<<<(144 * 160 + THR - 1) / THR, THR, 0, stream>>>(bas1t, bas2t);

  int nG = kRows * 128;
  stft_gather_kernel<<<(nG + THR - 1) / THR, THR, 0, stream>>>(x, frin);
  gemm_stft_kernel<<<(4104 * 9) / 8, THR, 0, stream>>>(frin, bas1t, Xbuf);
  gemm_conv1_kernel<<<(16 * 257) / 8, THR, 0, stream>>>(w1h, condp, b1, hbuf);
  fused_conv2_einsum_kernel<<<kB * 33, THR, 0, stream>>>(w2h, b2, hbuf, Xbuf, y16);
  gemm_istft_kernel<<<(4104 * 8) / 8, THR, 0, stream>>>(y16, bas2t, frout);
  int nOut = kB * kOut * kT;
  finalize_kernel<<<(nOut + THR - 1) / THR, THR, 0, stream>>>(frout, bias, out);
}
